// VisionMamba_67602785239636
// MI455X (gfx1250) — matrix operand order repaired and checked by reading, not compiled, not run on hardware
//
#include <hip/hip_runtime.h>

typedef __attribute__((ext_vector_type(16))) _Float16 v16h;
typedef __attribute__((ext_vector_type(8)))  _Float16 v8h;
typedef __attribute__((ext_vector_type(8)))  float    v8f;

#define DM     192
#define DI     384
#define DS     16
#define DTR    12
#define NPROJ  44      // DT_RANK + 2*D_STATE
#define LSEQ   197
#define BATCH  8
#define NLAYER 24
#define M_TOK  (BATCH*LSEQ)   // 1576
#define MP     1584           // padded to multiple of 16
#define NC     1000

// ---------------------------------------------------------------- utilities
__global__ void fill_zero_f32(float* __restrict__ p, int n) {
  int i = blockIdx.x*blockDim.x + threadIdx.x;
  if (i < n) p[i] = 0.f;
}
__global__ void fill_zero_f16(_Float16* __restrict__ p, int n) {
  int i = blockIdx.x*blockDim.x + threadIdx.x;
  if (i < n) p[i] = (_Float16)0.f;
}
__global__ void cvt_f32_to_f16(const float* __restrict__ s, _Float16* __restrict__ d, int n) {
  int i = blockIdx.x*blockDim.x + threadIdx.x;
  if (i < n) d[i] = (_Float16)s[i];
}

// ---------------------------------------------------------------- patch embed
// h[b,tok,:] = (tok==98 ? cls : conv_patch + patch_b) + pos_embed[tok]
__global__ __launch_bounds__(DM)
void patch_embed_kernel(const float* __restrict__ x, const float* __restrict__ pw,
                        const float* __restrict__ pb, const float* __restrict__ cls,
                        const float* __restrict__ pos, float* __restrict__ h)
{
  __shared__ float patch[768];
  int bt  = blockIdx.x;
  int b   = bt / LSEQ;
  int tok = bt % LSEQ;
  int d   = threadIdx.x;
  float out;
  if (tok == 98) {
    out = cls[d];
  } else {
    int p  = (tok < 98) ? tok : tok - 1;
    int py = p / 14, px = p % 14;
    for (int i = d; i < 768; i += DM) {
      int c = i >> 8; int rem = i & 255; int ky = rem >> 4; int kx = rem & 15;
      patch[i] = x[((size_t)(b*3 + c)*224 + (py*16 + ky))*224 + (px*16 + kx)];
    }
    __syncthreads();
    float acc = pb[d];
    const float* w = pw + (size_t)d*768;
    for (int i = 0; i < 768; ++i) acc += patch[i]*w[i];
    out = acc;
  }
  h[(size_t)bt*DM + d] = out + pos[tok*DM + d];
}

// ---------------------------------------------------------------- res += h; hn16 = rmsnorm(res)*w  (f16)
__global__ __launch_bounds__(DM)
void pre_rmsnorm_kernel(const float* __restrict__ h, float* __restrict__ res,
                        const float* __restrict__ nw, _Float16* __restrict__ hn16)
{
  __shared__ float red[256];
  int bt = blockIdx.x;
  int d  = threadIdx.x;
  float r = res[(size_t)bt*DM + d] + h[(size_t)bt*DM + d];
  res[(size_t)bt*DM + d] = r;
  if (d < 64) red[192 + d] = 0.f;
  red[d] = r*r;
  __syncthreads();
  for (int s = 128; s > 0; s >>= 1) { if (d < s) red[d] += red[d+s]; __syncthreads(); }
  float scale = rsqrtf(red[0]/(float)DM + 1e-5f);
  hn16[(size_t)bt*DM + d] = (_Float16)(r*scale*nw[d]);
}

// ---------------------------------------------------------------- WMMA GEMM  C[MP,N] = A[MP,K] * W[N,K]^T
// A,W f16 row-major; C f32 padded to MP rows (pad rows of A are zero -> pad C rows
// compute to zero, so the epilogue needs no bounds check: 8 straight-line stores).
template<int K>
__global__ __launch_bounds__(128)
void wmma_gemm_nt_kernel(const _Float16* __restrict__ A, const _Float16* __restrict__ W,
                         float* __restrict__ C, int ntM, int ntN, int ldc)
{
  int wave = threadIdx.x >> 5;
  int lane = threadIdx.x & 31;
  int tile = blockIdx.x*4 + wave;
  if (tile >= ntM*ntN) return;
  int tm = tile % ntM;
  int tn = tile / ntM;
  int hs = lane >> 4;       // lane half: 0 or 1
  int l16 = lane & 15;

  const _Float16* arow = A + (size_t)(tm*16 + l16)*K + hs*8;   // A 16x32 f16 layout
  const _Float16* brow = W + (size_t)(tn*16 + l16)*K + hs*8;   // B 32x16 f16 layout (same k order as A)

  v8f acc = {};
  #pragma unroll
  for (int k0 = 0; k0 < K; k0 += 32) {
    union { v16h v; v8h h[2]; } a;
    a.h[0] = *(const v8h*)(arow + k0);        // K k0..k0+7 (or +8..15)
    a.h[1] = *(const v8h*)(arow + k0 + 16);   // K k0+16..23 (or +24..31)
    union { v16h v; v8h h[2]; } bu;
    bu.h[0] = *(const v8h*)(brow + k0);
    bu.h[1] = *(const v8h*)(brow + k0 + 16);
    v16h bfrag = bu.v;
    acc = __builtin_amdgcn_wmma_f32_16x16x32_f16(false, a.v, false, bfrag,
                                                 (short)0, acc, false, false);
  }
  float* crow = C + (size_t)(tm*16 + hs*8)*ldc + tn*16 + l16;
  #pragma unroll
  for (int r = 0; r < 8; ++r) crow[(size_t)r*ldc] = acc[r];
}

// ---------------------------------------------------------------- causal conv1d + SiLU, both directions
__global__ void conv_silu_kernel(const float* __restrict__ xz,
                                 const float* __restrict__ cwf, const float* __restrict__ cbf,
                                 const float* __restrict__ cwb, const float* __restrict__ cbb,
                                 float* __restrict__ xc)
{
  int idx = blockIdx.x*blockDim.x + threadIdx.x;
  if (idx >= M_TOK*DI) return;
  int d  = idx % DI;
  int bt = idx / DI;
  int t  = bt % LSEQ;
  int b  = bt / LSEQ;
  float accf = cbf[d], accb = cbb[d];
  #pragma unroll
  for (int k = 0; k < 4; ++k) {
    int tt = t - 3 + k;
    if (tt >= 0) {
      accf += xz[((size_t)(b*LSEQ + tt))*768 + d]          * cwf[d*4 + k];
      accb += xz[((size_t)(b*LSEQ + (LSEQ-1-tt)))*768 + d] * cwb[d*4 + k];
    }
  }
  xc[idx]                    = accf * (1.f/(1.f + __expf(-accf)));
  xc[(size_t)M_TOK*DI + idx] = accb * (1.f/(1.f + __expf(-accb)));
}

// ---------------------------------------------------------------- x_proj (44x384 GEMV) + dt_proj + softplus
__global__ __launch_bounds__(64)
void xproj_dt_kernel(const float* __restrict__ xc,
                     const float* __restrict__ xwf, const float* __restrict__ xwb,
                     const float* __restrict__ dwf, const float* __restrict__ dbf,
                     const float* __restrict__ dwb, const float* __restrict__ dbb,
                     float* __restrict__ dbl, float* __restrict__ dt)
{
  __shared__ float sdbl[NPROJ];
  int bt  = blockIdx.x;
  int dir = blockIdx.y;
  int tid = threadIdx.x;
  const float* xrow = xc + (size_t)dir*M_TOK*DI + (size_t)bt*DI;
  const float* xw = dir ? xwb : xwf;
  const float* dw = dir ? dwb : dwf;
  const float* db = dir ? dbb : dbf;
  if (tid < NPROJ) {
    const float* wrow = xw + (size_t)tid*DI;
    float acc = 0.f;
    for (int k = 0; k < DI; ++k) acc += xrow[k]*wrow[k];
    sdbl[tid] = acc;
    dbl[((size_t)dir*M_TOK + bt)*NPROJ + tid] = acc;
  }
  __syncthreads();
  for (int d = tid; d < DI; d += 64) {
    float s = db[d];
    #pragma unroll
    for (int r = 0; r < DTR; ++r) s += sdbl[r]*dw[d*DTR + r];
    dt[((size_t)dir*M_TOK + bt)*DI + d] = (s > 20.f) ? s : log1pf(__expf(s));
  }
}

// ---------------------------------------------------------------- selective scan: 16 states/lane in VGPRs
__global__ __launch_bounds__(DI)
void scan_kernel(const float* __restrict__ xc, const float* __restrict__ dt,
                 const float* __restrict__ dbl,
                 const float* __restrict__ Alogf, const float* __restrict__ Alogb,
                 const float* __restrict__ Df,    const float* __restrict__ Db,
                 float* __restrict__ ys)
{
  __shared__ float sB[DS], sC[DS];
  int b   = blockIdx.x;
  int dir = blockIdx.y;
  int d   = threadIdx.x;
  const float* Alog = (dir ? Alogb : Alogf) + (size_t)d*DS;
  float Ad[DS], hst[DS];
  #pragma unroll
  for (int n = 0; n < DS; ++n) { Ad[n] = -__expf(Alog[n]); hst[n] = 0.f; }
  float Dp = (dir ? Db : Df)[d];
  size_t dbase = (size_t)dir*M_TOK*DI;
  size_t lbase = (size_t)dir*M_TOK*NPROJ;
  for (int t = 0; t < LSEQ; ++t) {
    int bt = b*LSEQ + t;
    if (d < 32) {
      float v = dbl[lbase + (size_t)bt*NPROJ + DTR + d];
      if (d < DS) sB[d] = v; else sC[d - DS] = v;
    }
    __syncthreads();
    float dtv = dt[dbase + (size_t)bt*DI + d];
    float xv  = xc[dbase + (size_t)bt*DI + d];
    float dtx = dtv*xv;
    float y = 0.f;
    #pragma unroll
    for (int n = 0; n < DS; ++n) {
      float a = __expf(dtv*Ad[n]);
      hst[n] = a*hst[n] + dtx*sB[n];
      y += hst[n]*sC[n];
    }
    ys[dbase + (size_t)bt*DI + d] = y + xv*Dp;
    __syncthreads();
  }
}

// ---------------------------------------------------------------- y = 0.5*silu(z)*(ys_f + rev(ys_b))  -> f16
__global__ void combine_silu_kernel(const float* __restrict__ ys, const float* __restrict__ xz,
                                    _Float16* __restrict__ y16)
{
  int idx = blockIdx.x*blockDim.x + threadIdx.x;
  if (idx >= M_TOK*DI) return;
  int d  = idx % DI;
  int bt = idx / DI;
  int t  = bt % LSEQ;
  int b  = bt / LSEQ;
  float yf = ys[idx];
  float yb = ys[(size_t)M_TOK*DI + (size_t)(b*LSEQ + (LSEQ-1-t))*DI + d];
  float z  = xz[(size_t)bt*768 + DI + d];
  float sz = z * (1.f/(1.f + __expf(-z)));
  y16[idx] = (_Float16)(0.5f*(yf + yb)*sz);
}

// ---------------------------------------------------------------- final norm (cls token only) + head
__global__ __launch_bounds__(DM)
void final_cls_kernel(const float* __restrict__ h, const float* __restrict__ res,
                      const float* __restrict__ nfw, float* __restrict__ cls)
{
  __shared__ float red[256];
  int b = blockIdx.x, d = threadIdx.x;
  int bt = b*LSEQ + 98;
  float r = res[(size_t)bt*DM + d] + h[(size_t)bt*DM + d];
  if (d < 64) red[192 + d] = 0.f;
  red[d] = r*r;
  __syncthreads();
  for (int s = 128; s > 0; s >>= 1) { if (d < s) red[d] += red[d+s]; __syncthreads(); }
  float scale = rsqrtf(red[0]/(float)DM + 1e-5f);
  cls[b*DM + d] = r*scale*nfw[d];
}

__global__ void head_kernel(const float* __restrict__ cls, const float* __restrict__ hw,
                            const float* __restrict__ hb, float* __restrict__ out)
{
  int b = blockIdx.x;
  int c = blockIdx.y*blockDim.x + threadIdx.x;
  if (c >= NC) return;
  float acc = hb[c];
  const float* xr = cls + (size_t)b*DM;
  const float* wr = hw + (size_t)c*DM;
  for (int k = 0; k < DM; ++k) acc += xr[k]*wr[k];
  out[b*NC + c] = acc;
}

// ---------------------------------------------------------------- driver
extern "C" void kernel_launch(void* const* d_in, const int* in_sizes, int n_in,
                              void* d_out, int out_size, void* d_ws, size_t ws_size,
                              hipStream_t stream)
{
  const float* x         = (const float*)d_in[0];
  const float* patch_w   = (const float*)d_in[1];
  const float* patch_b   = (const float*)d_in[2];
  const float* cls_tok   = (const float*)d_in[3];
  const float* pos_emb   = (const float*)d_in[4];
  const float* in_proj_w = (const float*)d_in[5];
  const float* conv_w    = (const float*)d_in[6];
  const float* conv_b    = (const float*)d_in[7];
  const float* x_proj_w  = (const float*)d_in[8];
  const float* dt_w      = (const float*)d_in[9];
  const float* dt_b      = (const float*)d_in[10];
  const float* A_log     = (const float*)d_in[11];
  const float* ssm_D     = (const float*)d_in[12];
  const float* conv_bw   = (const float*)d_in[13];
  const float* conv_bb   = (const float*)d_in[14];
  const float* x_proj_bw = (const float*)d_in[15];
  const float* dt_bw     = (const float*)d_in[16];
  const float* dt_bb     = (const float*)d_in[17];
  const float* A_blog    = (const float*)d_in[18];
  const float* ssm_Db    = (const float*)d_in[19];
  const float* out_proj_w= (const float*)d_in[20];
  const float* norm_w    = (const float*)d_in[21];
  const float* norm_f_w  = (const float*)d_in[22];
  const float* head_w    = (const float*)d_in[23];
  const float* head_b    = (const float*)d_in[24];
  float* out = (float*)d_out;
  (void)in_sizes; (void)n_in; (void)out_size; (void)ws_size;

  char* ws = (char*)d_ws;
  size_t off = 0;
  auto take = [&](size_t bytes) -> char* {
    char* p = ws + off;
    off = (off + bytes + 255) & ~(size_t)255;
    return p;
  };
  float*    f_h    = (float*)   take((size_t)MP*DM*4);     // padded rows: GEMM writes them
  float*    f_res  = (float*)   take((size_t)M_TOK*DM*4);
  _Float16* hn16   = (_Float16*)take((size_t)MP*DM*2);
  float*    f_xz   = (float*)   take((size_t)MP*768*4);    // padded rows: GEMM writes them
  float*    f_xc   = (float*)   take((size_t)2*M_TOK*DI*4);
  float*    f_dbl  = (float*)   take((size_t)2*M_TOK*NPROJ*4);
  float*    f_dt   = (float*)   take((size_t)2*M_TOK*DI*4);
  float*    f_ys   = (float*)   take((size_t)2*M_TOK*DI*4);
  _Float16* y16    = (_Float16*)take((size_t)MP*DI*2);
  _Float16* w16in  = (_Float16*)take((size_t)NLAYER*768*DM*2);
  _Float16* w16out = (_Float16*)take((size_t)NLAYER*DM*DI*2);
  float*    f_cls  = (float*)   take((size_t)BATCH*DM*4);

  int n;
  n = M_TOK*DM;
  fill_zero_f32<<<(n+255)/256, 256, 0, stream>>>(f_res, n);
  n = (MP - M_TOK)*DM;
  fill_zero_f16<<<(n+255)/256, 256, 0, stream>>>(hn16 + (size_t)M_TOK*DM, n);
  n = (MP - M_TOK)*DI;
  fill_zero_f16<<<(n+255)/256, 256, 0, stream>>>(y16 + (size_t)M_TOK*DI, n);
  n = NLAYER*768*DM;
  cvt_f32_to_f16<<<(n+255)/256, 256, 0, stream>>>(in_proj_w, w16in, n);
  n = NLAYER*DM*DI;
  cvt_f32_to_f16<<<(n+255)/256, 256, 0, stream>>>(out_proj_w, w16out, n);

  patch_embed_kernel<<<M_TOK, DM, 0, stream>>>(x, patch_w, patch_b, cls_tok, pos_emb, f_h);

  const int ntM = MP/16;        // 99
  for (int l = 0; l < NLAYER; ++l) {
    pre_rmsnorm_kernel<<<M_TOK, DM, 0, stream>>>(f_h, f_res, norm_w + l*DM, hn16);

    // xz[1584,768] = hn16[1584,192] @ in_proj_w[768,192]^T   (99*48 tiles, 4 waves/block)
    wmma_gemm_nt_kernel<DM><<<(ntM*48)/4, 128, 0, stream>>>(
        hn16, w16in + (size_t)l*768*DM, f_xz, ntM, 48, 768);

    n = M_TOK*DI;
    conv_silu_kernel<<<(n+255)/256, 256, 0, stream>>>(
        f_xz, conv_w + l*DI*4, conv_b + l*DI, conv_bw + l*DI*4, conv_bb + l*DI, f_xc);

    dim3 gx(M_TOK, 2);
    xproj_dt_kernel<<<gx, 64, 0, stream>>>(
        f_xc, x_proj_w + (size_t)l*NPROJ*DI, x_proj_bw + (size_t)l*NPROJ*DI,
        dt_w + (size_t)l*DI*DTR, dt_b + l*DI,
        dt_bw + (size_t)l*DI*DTR, dt_bb + l*DI, f_dbl, f_dt);

    dim3 gs(BATCH, 2);
    scan_kernel<<<gs, DI, 0, stream>>>(
        f_xc, f_dt, f_dbl,
        A_log + (size_t)l*DI*DS, A_blog + (size_t)l*DI*DS,
        ssm_D + l*DI, ssm_Db + l*DI, f_ys);

    combine_silu_kernel<<<(n+255)/256, 256, 0, stream>>>(f_ys, f_xz, y16);

    // h[1584,192] = y16[1584,384] @ out_proj_w[192,384]^T   (99*12 tiles)
    wmma_gemm_nt_kernel<DI><<<(ntM*12)/4, 128, 0, stream>>>(
        y16, w16out + (size_t)l*DM*DI, f_h, ntM, 12, DM);
  }

  final_cls_kernel<<<BATCH, DM, 0, stream>>>(f_h, f_res, norm_f_w, f_cls);
  dim3 gh(BATCH, (NC + 127)/128);
  head_kernel<<<gh, 128, 0, stream>>>(f_cls, head_w, head_b, out);
}
